// LSTM_39410619908490
// MI455X (gfx1250) — matrix operand order repaired and checked by reading, not compiled, not run on hardware
//
#include <hip/hip_runtime.h>

// ---------------------------------------------------------------------------
// 2-layer LSTM (H=100, L=4096, B=1) + final linear, fully on-chip on one WGP.
// Weights fp32 -> f16 into LDS once; per-wave constant B-fragments resident in
// VGPRs; recurrence = chained V_WMMA_F32_16X16X32_F16 (11 per timestep).
// A-fragments: inactive lanes read a zeroed LDS pad (no cndmask on the chain).
// Activations use fast hardware transcendentals (v_exp_f32 / v_rcp_f32).
// ---------------------------------------------------------------------------

typedef __attribute__((ext_vector_type(16))) _Float16 v16h;
typedef __attribute__((ext_vector_type(8)))  _Float16 v8h;
typedef __attribute__((ext_vector_type(8)))  float    v8f;

#define HID     100
#define SEQ     4096
#define NG      400            // 4*H gate outputs
#define NTILES  25             // NG / 16  (exact, no N padding)
#define NTHREADS (NTILES * 32) // 800 threads = 25 wave32 waves
#define KCAT    224            // layer-1 K: 200 real, padded to 7*32

// Branch-free fast activations on the v_exp_f32 / v_rcp_f32 path.
__device__ __forceinline__ float fast_sigmoid(float x) {
    return __builtin_amdgcn_rcpf(1.0f + __expf(-x));
}
__device__ __forceinline__ float fast_tanh(float x) {
    // tanh(x) = 1 - 2/(exp(2x)+1); saturates to +/-1 correctly for large |x|.
    return 1.0f - 2.0f * __builtin_amdgcn_rcpf(__expf(2.0f * x) + 1.0f);
}

// A fragment for V_WMMA_F32_16X16X32_F16: row M=0 = h[ktOff .. ktOff+31],
// all other rows zero.  ISA 16-bit A layout (7.12.2):
//   lanes 0-15  (M=lane)    : slots 0..7 -> K=0..7,  slots 8..15 -> K=16..23
//   lanes 16-31 (M=lane-16) : slots 0..7 -> K=8..15, slots 8..15 -> K=24..31
// `pa` is a per-lane pointer: active lanes (0,16) -> &h[hi]; others -> zero pad.
__device__ __forceinline__ v16h make_a_frag(const _Float16* __restrict__ pa,
                                            int ktOff) {
    union { v16h v; v8h p[2]; } u;
    u.p[0] = *(const v8h*)(pa + ktOff);        // slots 0..7
    u.p[1] = *(const v8h*)(pa + ktOff + 16);   // slots 8..15
    return u.v;
}

// B fragment (K=32 x N=16 tile of B[k][n] = W[k*NG + n], zero past Kreal).
// Dense 16-bit B layout: lanes 0-15: N=lane, slots 0..7 -> K=0..7, 8..15 -> K=16..23;
//                        lanes 16-31: N=lane-16, slots 0..7 -> K=8..15, 8..15 -> K=24..31.
__device__ __forceinline__ v16h make_b_frag(const _Float16* __restrict__ W,
                                            int Kreal, int ktOff, int n, int lane) {
    const int kb = ktOff + ((lane >= 16) ? 8 : 0);
    union { v16h v; _Float16 e[16]; } u;
#pragma unroll
    for (int s = 0; s < 16; ++s) {
        int k = kb + ((s < 8) ? s : s + 8);
        u.e[s] = (k < Kreal) ? W[k * NG + n] : (_Float16)0;
    }
    return u.v;
}

__global__ __launch_bounds__(NTHREADS, 1)
void lstm2_wmma_persistent(const float* __restrict__ x,
                           const float* __restrict__ Wih0, const float* __restrict__ Whh0,
                           const float* __restrict__ bih0, const float* __restrict__ bhh0,
                           const float* __restrict__ Wih1, const float* __restrict__ Whh1,
                           const float* __restrict__ bih1, const float* __restrict__ bhh1,
                           const float* __restrict__ Wlin, const float* __restrict__ blin,
                           float* __restrict__ out) {
    // ---- LDS: ~250 KB of the 320 KB WGP pool (single resident workgroup) ----
    __shared__ __align__(16) _Float16 Wt0[HID * NG];      // W_hh_l0^T        (K=100 x N=400)
    __shared__ __align__(16) _Float16 Wt1[2 * HID * NG];  // [W_ih_l1^T ; W_hh_l1^T] (K=200 x N=400)
    __shared__ __align__(16) _Float16 h1h[128];           // layer0 hidden, zero-padded K=128
    __shared__ __align__(16) _Float16 hcat[KCAT];         // [h1 ; h2 ; 0-pad]  layer-1 A vector
    __shared__ __align__(16) _Float16 zpad[KCAT];         // stays zero: feed to inactive A lanes
    __shared__ float bias0[NG], bias1[NG], wih0c[NG];
    __shared__ float gates[NG];
    __shared__ float c1s[HID], c2s[HID];

    const int tid  = threadIdx.x;
    const int lane = tid & 31;
    const int wv   = tid >> 5;      // 0..24 -> N-tile id

    // ---- prologue: transpose + fp32->f16 convert weights into LDS ----
    for (int idx = tid; idx < HID * NG; idx += NTHREADS) {
        int k = idx / NG, n = idx % NG;
        Wt0[idx]            = (_Float16)Whh0[n * HID + k];
        Wt1[idx]            = (_Float16)Wih1[n * HID + k];
        Wt1[HID * NG + idx] = (_Float16)Whh1[n * HID + k];
    }
    for (int n = tid; n < NG; n += NTHREADS) {
        bias0[n] = bih0[n] + bhh0[n];
        bias1[n] = bih1[n] + bhh1[n];
        wih0c[n] = Wih0[n];                 // W_ih_l0 is (400,1)
    }
    if (tid < 128)  h1h[tid] = (_Float16)0;
    if (tid < KCAT) { hcat[tid] = (_Float16)0; zpad[tid] = (_Float16)0; }
    if (tid < HID)  { c1s[tid] = 0.0f; c2s[tid] = 0.0f; }
    __syncthreads();

    // ---- per-lane A-source pointers: only lanes 0/16 carry M=0 data ----
    const int  hi  = (lane >= 16) ? 8 : 0;
    const bool act = ((lane & 15) == 0);
    const _Float16* pa1 = act ? (const _Float16*)&h1h[hi]  : (const _Float16*)&zpad[hi];
    const _Float16* pac = act ? (const _Float16*)&hcat[hi] : (const _Float16*)&zpad[hi];

    // ---- per-wave constant B fragments, resident in VGPRs for whole loop ----
    const int n0 = wv * 16;
    const int nc = n0 + (lane & 15);
    v16h B0[4], B1[7];
#pragma unroll
    for (int kt = 0; kt < 4; ++kt)
        B0[kt] = make_b_frag(Wt0, HID, kt * 32, nc, lane);        // Whh0 (x h1)
#pragma unroll
    for (int kt = 0; kt < 7; ++kt)
        B1[kt] = make_b_frag(Wt1, 2 * HID, kt * 32, nc, lane);    // [Wih1;Whh1] (x hcat)

    // ---- sequential recurrence: 4096 steps, everything stays on-chip ----
    for (int t = 0; t < SEQ; ++t) {
        const float xt = x[t];              // uniform -> scalar load

        // ----- layer 0 gates: Whh0 @ h1  (+ bias + W_ih0 * x_t) -----
        v8f acc = {};
#pragma unroll
        for (int kt = 0; kt < 4; ++kt)
            acc = __builtin_amdgcn_wmma_f32_16x16x32_f16(
                      false, make_a_frag(pa1, kt * 32), false, B0[kt],
                      (short)0, acc, false, false);
        if (lane < 16) {                    // row M=0 lives in acc[0], lanes 0-15
            int n = n0 + lane;
            gates[n] = acc[0] + bias0[n] + wih0c[n] * xt;
        }
        __syncthreads();

        // ----- layer 0 cell update (PyTorch gate order i,f,g,o) -----
        if (tid < HID) {
            float gi = fast_sigmoid(gates[tid]);
            float gf = fast_sigmoid(gates[HID + tid]);
            float gg = fast_tanh(gates[2 * HID + tid]);
            float go = fast_sigmoid(gates[3 * HID + tid]);
            float c  = gf * c1s[tid] + gi * gg;
            c1s[tid] = c;
            _Float16 hv = (_Float16)(go * fast_tanh(c));
            h1h[tid]  = hv;                 // layer-0 recurrent input
            hcat[tid] = hv;                 // layer-1 input, K = 0..99
        }
        __syncthreads();

        // ----- layer 1 gates: [Wih1;Whh1] @ [h1_t ; h2_{t-1}] + bias -----
        acc = (v8f){};
#pragma unroll
        for (int kt = 0; kt < 7; ++kt)
            acc = __builtin_amdgcn_wmma_f32_16x16x32_f16(
                      false, make_a_frag(pac, kt * 32), false, B1[kt],
                      (short)0, acc, false, false);
        if (lane < 16) {
            int n = n0 + lane;
            gates[n] = acc[0] + bias1[n];
        }
        __syncthreads();

        // ----- layer 1 cell update -----
        if (tid < HID) {
            float gi = fast_sigmoid(gates[tid]);
            float gf = fast_sigmoid(gates[HID + tid]);
            float gg = fast_tanh(gates[2 * HID + tid]);
            float go = fast_sigmoid(gates[3 * HID + tid]);
            float c  = gf * c2s[tid] + gi * gg;
            c2s[tid] = c;
            hcat[HID + tid] = (_Float16)(go * fast_tanh(c));   // h2, K = 100..199
        }
        __syncthreads();
    }

    // ---- final linear on last h2 ----
    if (tid == 0) {
        float s = blin[0];
#pragma unroll 4
        for (int u = 0; u < HID; ++u) s += Wlin[u] * (float)hcat[HID + u];
        out[0] = s;
    }
}

extern "C" void kernel_launch(void* const* d_in, const int* in_sizes, int n_in,
                              void* d_out, int out_size, void* d_ws, size_t ws_size,
                              hipStream_t stream) {
    (void)in_sizes; (void)n_in; (void)out_size; (void)d_ws; (void)ws_size;
    const float* x    = (const float*)d_in[0];
    const float* Wih0 = (const float*)d_in[1];
    const float* Whh0 = (const float*)d_in[2];
    const float* bih0 = (const float*)d_in[3];
    const float* bhh0 = (const float*)d_in[4];
    const float* Wih1 = (const float*)d_in[5];
    const float* Whh1 = (const float*)d_in[6];
    const float* bih1 = (const float*)d_in[7];
    const float* bhh1 = (const float*)d_in[8];
    const float* Wlin = (const float*)d_in[9];
    const float* blin = (const float*)d_in[10];
    float* out = (float*)d_out;

    lstm2_wmma_persistent<<<1, NTHREADS, 0, stream>>>(
        x, Wih0, Whh0, bih0, bhh0, Wih1, Whh1, bih1, bhh1, Wlin, blin, out);
}
